// EvoformerStack_36275293782044
// MI455X (gfx1250) — matrix operand order repaired and checked by reading, not compiled, not run on hardware
//
#include <hip/hip_runtime.h>
#include <cmath>

// ---------------- CDNA5 WMMA vector types ----------------
typedef __attribute__((ext_vector_type(16))) __bf16 v16bf;
typedef __attribute__((ext_vector_type(4)))  __bf16 v4bf;
typedef __attribute__((ext_vector_type(8)))  float  v8f;

// ---------------- model dims ----------------
#define NRES  256
#define NN    (NRES * NRES)          // 65536
#define CM    256
#define CZ    128
#define HSEQ  8
#define HPAIR 4
#define CHID  32
#define CS    384
#define BIGF  ((size_t)NN * CZ)      // 8,388,608 floats = 32 MB
static const float QSCALE = 0.17677669529663687f;   // 32^-0.5

// =====================================================================
// Generic batched-strided GEMM:  C = act(alpha*(A@B(^T) + bias)) [+ C]
// bf16 WMMA 16x16x32, fp32 accumulate.
// Tile 64x64, 4 waves (wave32), each wave owns a 32x32 block (4 WMMAs).
// - LDS tiles are pre-swizzled to WMMA fragment order: each lane reads its
//   16-element fragment as one aligned 32B (2x ds_load_b128).
// - Software pipelined: tile k+1 is loaded to registers while tile k's
//   WMMAs run; staging pointers hoisted out of the K-loop.
// - transB is a template parameter; N-tail handled by pointer clamp +
//   a loop-invariant 0/1 multiplier (no exec-mask divergence).
// Assumes M % 64 == 0 and K % 32 == 0 (true at every call site).
// =====================================================================
#define TM 64
#define TN 64
#define TK 32

template <int TRB>
__global__ __launch_bounds__(128)
void k_gemm_t(const float* __restrict__ Ag, long long lda, long long sA0, long long sA1,
              const float* __restrict__ Bg, long long ldb, long long sB0, long long sB1,
              const float* __restrict__ bias,
              float* __restrict__ Cg, long long ldc, long long sC0, long long sC1,
              int M, int N, int K, int nb0, float alpha, int act, int beta)
{
    // fragment-order tiles: [16-row strip / 16-col frag][lane][element]
    __shared__ __align__(32) __bf16 AsF[4][32][16];
    __shared__ __align__(32) __bf16 BsF[4][32][16];

    const int tid  = threadIdx.x;
    const int lane = tid & 31;
    const int wv   = tid >> 5;          // wave 0..3
    const int wm   = wv >> 1;           // row block wm*32
    const int wn   = wv & 1;            // col block wn*32
    const int half = lane >> 4;
    const int l16  = lane & 15;

    const int bz = (int)blockIdx.z;
    const int b0 = bz % nb0;
    const int b1 = bz / nb0;
    const float* Ab = Ag + (long long)b0 * sA0 + (long long)b1 * sA1;
    const float* Bb = Bg + (long long)b0 * sB0 + (long long)b1 * sB1;
    float*       Cb = Cg + (long long)b0 * sC0 + (long long)b1 * sC1;

    const int m0 = (int)blockIdx.y * TM;
    const int n0 = (int)blockIdx.x * TN;

    // ---- loop-invariant staging descriptors ----
    const float* pa[4]; v4bf* sa[4];
    const float* pb[4]; v4bf* sb[4]; float bmul[4];
#pragma unroll
    for (int t = 0; t < 4; ++t) {
        int u = tid + t * 128;          // 0..511
        {   // A unit: tile row r, k-quad c4
            int r = u >> 3, c4 = (u & 7) * 4;
            pa[t] = Ab + (long long)(m0 + r) * lda + c4;
            // ISA A-frag: lane = (row&15) + 16*((k>>3)&1), e = ((k>>4)<<3)+(k&7)
            int strip = r >> 4;
            int ln    = (r & 15) + (((c4 >> 3) & 1) << 4);
            int e0    = ((c4 >> 4) << 3) + (c4 & 7);
            sa[t] = (v4bf*)&AsF[strip][ln][e0];
        }
        {   // B unit: tile col c, k-quad q
            int c = u >> 3, q = u & 7;
            int gn = n0 + c;
            bool ok = (gn < N);
            bmul[t] = ok ? 1.f : 0.f;   // branch-free tail zeroing
            int gnc = ok ? gn : 0;      // clamp: always a valid address
            if (TRB) pb[t] = Bb + (long long)gnc * ldb + q * 4;
            else     pb[t] = Bb + (long long)(q * 4) * ldb + gnc;
            // ISA B-frag: lane = (col&15) + 16*((k>>3)&1), e = ((k>>4)<<3)+(k&7)
            int nf = c >> 4;
            int ln = (c & 15) + (((q >> 1) & 1) << 4);
            int e0 = ((q >> 2) << 3) + (q & 1) * 4;
            sb[t] = (v4bf*)&BsF[nf][ln][e0];
        }
    }

    v8f acc00 = {}, acc01 = {}, acc10 = {}, acc11 = {};
    float4 fa[4], fb[4];

    // prologue: load tile 0
#pragma unroll
    for (int t = 0; t < 4; ++t) fa[t] = *(const float4*)(pa[t]);
#pragma unroll
    for (int t = 0; t < 4; ++t) {
        if (TRB) fb[t] = *(const float4*)(pb[t]);
        else {
            const float* p = pb[t];
            fb[t].x = p[0]; fb[t].y = p[ldb]; fb[t].z = p[2 * ldb]; fb[t].w = p[3 * ldb];
        }
    }

    for (int k0 = 0; k0 < K; k0 += TK) {
        // ---- convert current tile registers -> fragment-order LDS ----
#pragma unroll
        for (int t = 0; t < 4; ++t) {
            v4bf h = { (__bf16)fa[t].x, (__bf16)fa[t].y, (__bf16)fa[t].z, (__bf16)fa[t].w };
            *sa[t] = h;
        }
#pragma unroll
        for (int t = 0; t < 4; ++t) {
            v4bf h = { (__bf16)(fb[t].x * bmul[t]), (__bf16)(fb[t].y * bmul[t]),
                       (__bf16)(fb[t].z * bmul[t]), (__bf16)(fb[t].w * bmul[t]) };
            *sb[t] = h;
        }
        __syncthreads();

        // ---- pipeline: issue next tile's global loads now ----
        if (k0 + TK < K) {
            long long kA = k0 + TK;
            long long kB = (long long)(k0 + TK) * ldb;
#pragma unroll
            for (int t = 0; t < 4; ++t) fa[t] = *(const float4*)(pa[t] + kA);
#pragma unroll
            for (int t = 0; t < 4; ++t) {
                if (TRB) fb[t] = *(const float4*)(pb[t] + kA);
                else {
                    const float* p = pb[t] + kB;
                    fb[t].x = p[0]; fb[t].y = p[ldb]; fb[t].z = p[2 * ldb]; fb[t].w = p[3 * ldb];
                }
            }
            // deepen the pipeline: hint tile k+2 toward cache (global_prefetch_b8)
            if (k0 + 2 * TK < K)
                __builtin_prefetch(pa[0] + k0 + 2 * TK, 0, 1);
        }

        // ---- one aligned 32B read per fragment, then 4 WMMAs ----
        v16bf a0 = *(const v16bf*)&AsF[wm * 2 + 0][lane][0];
        v16bf a1 = *(const v16bf*)&AsF[wm * 2 + 1][lane][0];
        v16bf b0 = *(const v16bf*)&BsF[wn * 2 + 0][lane][0];
        v16bf b1 = *(const v16bf*)&BsF[wn * 2 + 1][lane][0];

        acc00 = __builtin_amdgcn_wmma_f32_16x16x32_bf16(false, a0, false, b0, (short)0, acc00, false, false);
        acc01 = __builtin_amdgcn_wmma_f32_16x16x32_bf16(false, a0, false, b1, (short)0, acc01, false, false);
        acc10 = __builtin_amdgcn_wmma_f32_16x16x32_bf16(false, a1, false, b0, (short)0, acc10, false, false);
        acc11 = __builtin_amdgcn_wmma_f32_16x16x32_bf16(false, a1, false, b1, (short)0, acc11, false, false);
        __syncthreads();
    }

    // ---- epilogue: D layout -> M = r + 8*half, N = lane&15 ----
#pragma unroll
    for (int s = 0; s < 2; ++s) {
#pragma unroll
        for (int f = 0; f < 2; ++f) {
            v8f acc = (s == 0) ? (f == 0 ? acc00 : acc01)
                               : (f == 0 ? acc10 : acc11);
            int n = n0 + (wn * 2 + f) * 16 + l16;
            if (n >= N) continue;
            int mbase = m0 + (wm * 2 + s) * 16 + half * 8;
            float bv = bias ? bias[n] : 0.f;
#pragma unroll
            for (int r = 0; r < 8; ++r) {
                float v = (acc[r] + bv) * alpha;
                if (act == 1)      v = fmaxf(v, 0.f);
                else if (act == 2) v = 1.f / (1.f + __expf(-v));
                long long off = (long long)(mbase + r) * ldc + n;
                if (beta) v += Cb[off];
                Cb[off] = v;
            }
        }
    }
}

// =====================================================================
// LayerNorm: one wave32 per row (wave shuffle reductions)
// =====================================================================
__global__ __launch_bounds__(256)
void k_layernorm(const float* __restrict__ x, const float* __restrict__ g,
                 const float* __restrict__ b, float* __restrict__ y,
                 long long rows, int C)
{
    long long row = (long long)blockIdx.x * 8 + (threadIdx.x >> 5);
    if (row >= rows) return;
    int lane = threadIdx.x & 31;
    const float* xr = x + row * C;
    float s = 0.f;
    for (int c = lane; c < C; c += 32) s += xr[c];
#pragma unroll
    for (int o = 16; o; o >>= 1) s += __shfl_xor(s, o, 32);
    float mu = s / (float)C;
    float vv = 0.f;
    for (int c = lane; c < C; c += 32) { float d = xr[c] - mu; vv += d * d; }
#pragma unroll
    for (int o = 16; o; o >>= 1) vv += __shfl_xor(vv, o, 32);
    float inv = rsqrtf(vv / (float)C + 1e-5f);
    float* yr = y + row * C;
    for (int c = lane; c < C; c += 32) yr[c] = (xr[c] - mu) * inv * g[c] + b[c];
}

// =====================================================================
// Softmax kernels (K dim fixed at 256 == blockDim)
// =====================================================================
__global__ __launch_bounds__(256)
void k_seq_softmax(float* __restrict__ lg, const float* __restrict__ pb,
                   const float* __restrict__ sm)
{
    int row = blockIdx.x;            // row = h*256 + q
    int q = row & 255, h = row >> 8;
    int k = threadIdx.x;
    __shared__ float sh[256];
    float x = lg[(size_t)row * 256 + k]
            + pb[((size_t)q * 256 + k) * HSEQ + h]
            + (sm[k] - 1.f) * 1e9f;
    sh[k] = x; __syncthreads();
    for (int s = 128; s; s >>= 1) { if (k < s) sh[k] = fmaxf(sh[k], sh[k + s]); __syncthreads(); }
    float mx = sh[0]; __syncthreads();
    float e = __expf(x - mx); sh[k] = e; __syncthreads();
    for (int s = 128; s; s >>= 1) { if (k < s) sh[k] += sh[k + s]; __syncthreads(); }
    lg[(size_t)row * 256 + k] = e / sh[0];
}

__global__ __launch_bounds__(256)
void k_tri_softmax(float* __restrict__ lg, const float* __restrict__ tb,
                   const float* __restrict__ pm, int i0)
{
    int row = blockIdx.x;            // row = ((ic*4)+h)*256 + q
    int q  = row & 255;
    int h  = (row >> 8) & 3;
    int ic = row >> 10;
    int i  = i0 + ic;
    int k = threadIdx.x;
    __shared__ float sh[256];
    float x = lg[(size_t)row * 256 + k]
            + tb[((size_t)q * 256 + k) * HPAIR + h]
            + (pm[(size_t)i * 256 + k] - 1.f) * 1e9f;
    sh[k] = x; __syncthreads();
    for (int s = 128; s; s >>= 1) { if (k < s) sh[k] = fmaxf(sh[k], sh[k + s]); __syncthreads(); }
    float mx = sh[0]; __syncthreads();
    float e = __expf(x - mx); sh[k] = e; __syncthreads();
    for (int s = 128; s; s >>= 1) { if (k < s) sh[k] += sh[k + s]; __syncthreads(); }
    lg[(size_t)row * 256 + k] = e / sh[0];
}

// =====================================================================
// Elementwise / permute kernels (grid-stride)
// =====================================================================
#define GS_LOOP(i, n) for (size_t i = (size_t)blockIdx.x * blockDim.x + threadIdx.x; \
                           i < (n); i += (size_t)gridDim.x * blockDim.x)

__global__ void k_mul(float* a, const float* b, size_t n) { GS_LOOP(i, n) a[i] *= b[i]; }

__global__ void k_rowmask_mul(float* a, const float* mask, size_t n, int C)
{ GS_LOOP(i, n) a[i] *= mask[i / C]; }

__global__ void k_residual_rowmask(float* dst, const float* src, const float* mask,
                                   size_t n, int C)
{ GS_LOOP(i, n) dst[i] += src[i] * mask[i / C]; }

__global__ void k_trimul_combine(float* p, const float* g, const float* mask,
                                 size_t n, int C)
{ GS_LOOP(i, n) p[i] = mask[i / C] * p[i] * g[i]; }

__global__ void k_gated_residual(float* z, const float* g, const float* o, size_t n)
{ GS_LOOP(i, n) z[i] += g[i] * o[i]; }

__global__ void k_opm_finalize(float* z, const float* outer, const float* sm, size_t n)
{
    GS_LOOP(i, n) {
        size_t ij = i / CZ;
        int jj = (int)(ij & (NRES - 1));
        int ii = (int)(ij >> 8);
        z[i] += outer[i] / (sm[ii] * sm[jj] + 1e-3f);
    }
}

// dst[c][i][k] = src[(i,k or k,i)][c]   (channel-major repack for trimul)
__global__ void k_nc_to_cnk(float* dst, const float* src, int swap, size_t n)
{
    GS_LOOP(i, n) {
        int k = (int)(i & (NRES - 1));
        int r = (int)((i >> 8) & (NRES - 1));
        int c = (int)(i >> 16);
        size_t s = swap ? ((size_t)k * NRES + r) : ((size_t)r * NRES + k);
        dst[i] = src[s * CZ + c];
    }
}

// dst[i][j][c] = src[c][i][j]
__global__ void k_cnk_to_nc(float* dst, const float* src, size_t n)
{
    GS_LOOP(i, n) {
        int c = (int)(i & (CZ - 1));
        size_t ij = i / CZ;
        int j = (int)(ij & (NRES - 1));
        int r = (int)(ij >> 8);
        dst[i] = src[(size_t)c * NN + (size_t)r * NRES + j];
    }
}

// dst[(j,i)][c] = src[(i,j)][c]
__global__ void k_transpose_pair(float* dst, const float* src, size_t n)
{
    GS_LOOP(i, n) {
        int c = (int)(i & (CZ - 1));
        size_t ij = i / CZ;
        int jj = (int)(ij & (NRES - 1));
        int ii = (int)(ij >> 8);
        dst[i] = src[((size_t)jj * NRES + ii) * CZ + c];
    }
}

__global__ void k_transpose_mask(float* dst, const float* src, size_t n)
{
    GS_LOOP(i, n) {
        int jj = (int)(i & (NRES - 1));
        int ii = (int)(i >> 8);
        dst[i] = src[(size_t)jj * NRES + ii];
    }
}

// z[(i,j)][c] += src[(j,i)][c]
__global__ void k_add_transposed(float* z, const float* src, size_t n)
{
    GS_LOOP(i, n) {
        int c = (int)(i & (CZ - 1));
        size_t ij = i / CZ;
        int jj = (int)(ij & (NRES - 1));
        int ii = (int)(ij >> 8);
        z[i] += src[((size_t)jj * NRES + ii) * CZ + c];
    }
}

// =====================================================================
// Host-side parameter unpacking (JAX pytree sorted-key flattening)
// =====================================================================
struct Lin  { const float *w, *b; };
struct LNp  { const float *g, *b; };
struct AttP { LNp ln, ln_z; Lin q, k, v, g, o; const float* b_w; };
struct TransP { LNp ln; Lin l1, l2; };
struct TriP { LNp ln_in, ln_out; Lin ap, ag, bp, bg, g, out; };
struct OpmP { LNp ln; Lin l1, l2, out; };
struct BlockP { AttP seq_att; TransP seq_trans; OpmP opm; TriP tmo, tmi;
                AttP tas, tae; TransP pair_trans; };

static const float* nx(void* const* d, int& i) { return (const float*)d[i++]; }
static void fLin(void* const* d, int& i, Lin& l)  { l.b = nx(d, i); l.w = nx(d, i); }   // "b"<"w"
static void fLN (void* const* d, int& i, LNp& l)  { l.b = nx(d, i); l.g = nx(d, i); }   // "b"<"g"
static void fAtt(void* const* d, int& i, AttP& a, bool lnz) {
    a.b_w = nx(d, i); fLin(d, i, a.g); fLin(d, i, a.k); fLN(d, i, a.ln);
    if (lnz) fLN(d, i, a.ln_z);
    fLin(d, i, a.o); fLin(d, i, a.q); fLin(d, i, a.v);
}
static void fTrans(void* const* d, int& i, TransP& t) { fLin(d, i, t.l1); fLin(d, i, t.l2); fLN(d, i, t.ln); }
static void fTri(void* const* d, int& i, TriP& t) {
    fLin(d, i, t.ag); fLin(d, i, t.ap); fLin(d, i, t.bg); fLin(d, i, t.bp);
    fLin(d, i, t.g);  fLN(d, i, t.ln_in); fLN(d, i, t.ln_out); fLin(d, i, t.out);
}
static void fOpm(void* const* d, int& i, OpmP& o) { fLin(d, i, o.l1); fLin(d, i, o.l2); fLN(d, i, o.ln); fLin(d, i, o.out); }
static void fBlock(void* const* d, int& i, BlockP& b) {
    fOpm(d, i, b.opm); fTrans(d, i, b.pair_trans); fAtt(d, i, b.seq_att, true);
    fTrans(d, i, b.seq_trans); fAtt(d, i, b.tae, false); fAtt(d, i, b.tas, false);
    fTri(d, i, b.tmi); fTri(d, i, b.tmo);
}

// =====================================================================
// Host launch helpers
// =====================================================================
static inline dim3 ewg(size_t n) {
    size_t b = (n + 255) / 256; if (b > 16384) b = 16384; return dim3((unsigned)b);
}

static void gemm(hipStream_t st,
                 const float* A, long long lda, long long sA0, long long sA1,
                 const float* B, long long ldb, long long sB0, long long sB1, int transB,
                 const float* bias,
                 float* C, long long ldc, long long sC0, long long sC1,
                 int M, int N, int K, int nb0, int nb1,
                 float alpha = 1.f, int act = 0, int beta = 0)
{
    dim3 g((N + TN - 1) / TN, (M + TM - 1) / TM, nb0 * nb1);
    if (transB)
        k_gemm_t<1><<<g, 128, 0, st>>>(A, lda, sA0, sA1, B, ldb, sB0, sB1, bias,
                                       C, ldc, sC0, sC1, M, N, K, nb0, alpha, act, beta);
    else
        k_gemm_t<0><<<g, 128, 0, st>>>(A, lda, sA0, sA1, B, ldb, sB0, sB1, bias,
                                       C, ldc, sC0, sC1, M, N, K, nb0, alpha, act, beta);
}

static void lnrun(hipStream_t st, const float* x, const LNp& p, float* y,
                  size_t rows, int C)
{
    dim3 g((unsigned)((rows + 7) / 8));
    k_layernorm<<<g, 256, 0, st>>>(x, p.g, p.b, y, (long long)rows, C);
}

// ---------------- composite ops ----------------
static void seq_attention(hipStream_t st, float* m, float* z, const float* sm,
                          const AttP& p, float* m_ln, float* zln, float* q, float* kk,
                          float* v, float* g, float* pb, float* lg, float* o)
{
    lnrun(st, m, p.ln, m_ln, NRES, CM);
    lnrun(st, z, p.ln_z, zln, NN, CZ);
    gemm(st, m_ln, CM, 0, 0, p.q.w, CM, 0, 0, 0, p.q.b, q,  CM, 0, 0, NRES, CM, CM, 1, 1, QSCALE, 0, 0);
    gemm(st, m_ln, CM, 0, 0, p.k.w, CM, 0, 0, 0, p.k.b, kk, CM, 0, 0, NRES, CM, CM, 1, 1);
    gemm(st, m_ln, CM, 0, 0, p.v.w, CM, 0, 0, 0, p.v.b, v,  CM, 0, 0, NRES, CM, CM, 1, 1);
    gemm(st, m_ln, CM, 0, 0, p.g.w, CM, 0, 0, 0, p.g.b, g,  CM, 0, 0, NRES, CM, CM, 1, 1, 1.f, 2, 0);
    gemm(st, zln, CZ, 0, 0, p.b_w, HSEQ, 0, 0, 0, nullptr, pb, HSEQ, 0, 0, NN, HSEQ, CZ, 1, 1);
    // logits[h,q,k] = Q_h @ K_h^T
    gemm(st, q, CM, CHID, 0, kk, CM, CHID, 0, 1, nullptr, lg, NRES, (long long)NN, 0,
         NRES, NRES, CHID, HSEQ, 1);
    k_seq_softmax<<<dim3(HSEQ * NRES), 256, 0, st>>>(lg, pb, sm);
    // o[q,h,c] = A_h @ V_h
    gemm(st, lg, NRES, (long long)NN, 0, v, CM, CHID, 0, 0, nullptr, o, CM, CHID, 0,
         NRES, CHID, NRES, HSEQ, 1);
    k_mul<<<ewg(NN), 256, 0, st>>>(o, g, (size_t)NN);
    gemm(st, o, CM, 0, 0, p.o.w, CM, 0, 0, 0, p.o.b, m, CM, 0, 0, NRES, CM, CM, 1, 1, 1.f, 0, 1);
}

static void trimul(hipStream_t st, float* z, const float* pm, const TriP& p, bool outgoing,
                   float* bA, float* bB, float* bC, float* bD, float* bE, float* bG)
{
    lnrun(st, z, p.ln_in, bA, NN, CZ);
    gemm(st, bA, CZ, 0, 0, p.ap.w, CZ, 0, 0, 0, p.ap.b, bB, CZ, 0, 0, NN, CZ, CZ, 1, 1);
    gemm(st, bA, CZ, 0, 0, p.ag.w, CZ, 0, 0, 0, p.ag.b, bC, CZ, 0, 0, NN, CZ, CZ, 1, 1, 1.f, 2, 0);
    k_trimul_combine<<<ewg(BIGF), 256, 0, st>>>(bB, bC, pm, BIGF, CZ);
    gemm(st, bA, CZ, 0, 0, p.bp.w, CZ, 0, 0, 0, p.bp.b, bC, CZ, 0, 0, NN, CZ, CZ, 1, 1);
    gemm(st, bA, CZ, 0, 0, p.bg.w, CZ, 0, 0, 0, p.bg.b, bD, CZ, 0, 0, NN, CZ, CZ, 1, 1, 1.f, 2, 0);
    k_trimul_combine<<<ewg(BIGF), 256, 0, st>>>(bC, bD, pm, BIGF, CZ);
    // channel-major repack (swap i/k for incoming)
    int swap = outgoing ? 0 : 1;
    k_nc_to_cnk<<<ewg(BIGF), 256, 0, st>>>(bD, bB, swap, BIGF);
    k_nc_to_cnk<<<ewg(BIGF), 256, 0, st>>>(bE, bC, swap, BIGF);
    // X_c = A'_c @ B'_c^T, 128 batched 256x256x256 GEMMs
    gemm(st, bD, NRES, (long long)NN, 0, bE, NRES, (long long)NN, 0, 1, nullptr,
         bG, NRES, (long long)NN, 0, NRES, NRES, NRES, CZ, 1);
    k_cnk_to_nc<<<ewg(BIGF), 256, 0, st>>>(bB, bG, BIGF);
    lnrun(st, bB, p.ln_out, bC, NN, CZ);
    gemm(st, bA, CZ, 0, 0, p.g.w, CZ, 0, 0, 0, p.g.b, bD, CZ, 0, 0, NN, CZ, CZ, 1, 1, 1.f, 2, 0);
    gemm(st, bC, CZ, 0, 0, p.out.w, CZ, 0, 0, 0, p.out.b, bE, CZ, 0, 0, NN, CZ, CZ, 1, 1);
    k_gated_residual<<<ewg(BIGF), 256, 0, st>>>(z, bD, bE, BIGF);
}

static void tri_attention(hipStream_t st, float* z, const float* zin, const float* pm,
                          const AttP& p, float* bA, float* bB, float* bC, float* bD,
                          float* bE, float* bF, float* bG, float* tb, bool transpose_out)
{
    lnrun(st, zin, p.ln, bA, NN, CZ);
    gemm(st, bA, CZ, 0, 0, p.q.w, CZ, 0, 0, 0, p.q.b, bB, CZ, 0, 0, NN, CZ, CZ, 1, 1, QSCALE, 0, 0);
    gemm(st, bA, CZ, 0, 0, p.k.w, CZ, 0, 0, 0, p.k.b, bC, CZ, 0, 0, NN, CZ, CZ, 1, 1);
    gemm(st, bA, CZ, 0, 0, p.v.w, CZ, 0, 0, 0, p.v.b, bD, CZ, 0, 0, NN, CZ, CZ, 1, 1);
    gemm(st, bA, CZ, 0, 0, p.g.w, CZ, 0, 0, 0, p.g.b, bE, CZ, 0, 0, NN, CZ, CZ, 1, 1, 1.f, 2, 0);
    gemm(st, bA, CZ, 0, 0, p.b_w, HPAIR, 0, 0, 0, nullptr, tb, HPAIR, 0, 0, NN, HPAIR, CZ, 1, 1);

    const long long ROW = (long long)NRES * CZ;     // 32768
    for (int i0 = 0; i0 < NRES; i0 += 32) {
        long long off = (long long)i0 * ROW;
        // logits[ic,h,q,k] = Q @ K^T   (batch: h stride 32, ic stride ROW)
        gemm(st, bB + off, CZ, CHID, ROW, bC + off, CZ, CHID, ROW, 1, nullptr,
             bG, NRES, (long long)NN, 4LL * NN, NRES, NRES, CHID, HPAIR, 32);
        k_tri_softmax<<<dim3(32 * HPAIR * NRES), 256, 0, st>>>(bG, tb, pm, i0);
        // o = A @ V
        gemm(st, bG, NRES, (long long)NN, 4LL * NN, bD + off, CZ, CHID, ROW, 0, nullptr,
             bF + off, CZ, CHID, ROW, NRES, CHID, NRES, HPAIR, 32);
    }
    k_mul<<<ewg(BIGF), 256, 0, st>>>(bF, bE, BIGF);
    if (!transpose_out) {
        gemm(st, bF, CZ, 0, 0, p.o.w, CZ, 0, 0, 0, p.o.b, z, CZ, 0, 0, NN, CZ, CZ, 1, 1, 1.f, 0, 1);
    } else {
        gemm(st, bF, CZ, 0, 0, p.o.w, CZ, 0, 0, 0, p.o.b, bB, CZ, 0, 0, NN, CZ, CZ, 1, 1);
        k_add_transposed<<<ewg(BIGF), 256, 0, st>>>(z, bB, BIGF);
    }
}

// =====================================================================
// kernel_launch
// =====================================================================
extern "C" void kernel_launch(void* const* d_in, const int* in_sizes, int n_in,
                              void* d_out, int out_size, void* d_ws, size_t ws_size,
                              hipStream_t stream)
{
    (void)in_sizes; (void)n_in; (void)out_size; (void)ws_size;

    const float* sm = (const float*)d_in[2];       // seq_mask  [256]
    const float* pm = (const float*)d_in[3];       // pair_mask [256,256]

    // params
    int idx = 5;
    BlockP blk[2];
    fBlock(d_in, idx, blk[0]);
    fBlock(d_in, idx, blk[1]);
    const float* fin_b = nx(d_in, idx);
    const float* fin_w = nx(d_in, idx);

    // outputs (m, z, s concatenated); m and z evolve in place
    float* out_m = (float*)d_out;
    float* out_z = out_m + NN;
    float* out_s = out_z + BIGF;

    // workspace: 8 big z-sized buffers + small pool
    float* ws = (float*)d_ws;
    float* bA = ws;
    float* bB = bA + BIGF; float* bC = bB + BIGF; float* bD = bC + BIGF;
    float* bE = bD + BIGF; float* bF = bE + BIGF; float* bG = bF + BIGF;
    float* bH = bG + BIGF;
    float* sp = bH + BIGF;
    float* m_ln   = sp; sp += NN;
    float* s_q    = sp; sp += NN;
    float* s_k    = sp; sp += NN;
    float* s_v    = sp; sp += NN;
    float* s_g    = sp; sp += NN;
    float* pb     = sp; sp += (size_t)NN * HSEQ;
    float* s_lg   = sp; sp += (size_t)HSEQ * NN;
    float* s_o    = sp; sp += NN;
    float* s_t1   = sp; sp += (size_t)NRES * 1024;
    float* s_tmp  = sp; sp += NN;
    float* a_op   = sp; sp += NRES * 32;
    float* b_op   = sp; sp += NRES * 32;
    float* T_op   = sp; sp += (size_t)32 * NRES * CZ;
    float* tb     = sp; sp += (size_t)NN * HPAIR;
    float* pm_t   = sp; sp += NN;

    hipMemcpyAsync(out_m, d_in[0], (size_t)NN * 4, hipMemcpyDeviceToDevice, stream);
    hipMemcpyAsync(out_z, d_in[1], BIGF * 4, hipMemcpyDeviceToDevice, stream);

    for (int bi = 0; bi < 2; ++bi) {
        const BlockP& P = blk[bi];

        // ---- m = m + seq_attention ----
        seq_attention(stream, out_m, out_z, sm, P.seq_att,
                      m_ln, bA, s_q, s_k, s_v, s_g, pb, s_lg, s_o);

        // ---- m = m + transition(m) ----
        lnrun(stream, out_m, P.seq_trans.ln, m_ln, NRES, CM);
        gemm(stream, m_ln, CM, 0, 0, P.seq_trans.l1.w, 1024, 0, 0, 0, P.seq_trans.l1.b,
             s_t1, 1024, 0, 0, NRES, 1024, CM, 1, 1, 1.f, 1, 0);
        gemm(stream, s_t1, 1024, 0, 0, P.seq_trans.l2.w, CM, 0, 0, 0, P.seq_trans.l2.b,
             s_tmp, CM, 0, 0, NRES, CM, 1024, 1, 1);
        k_residual_rowmask<<<ewg(NN), 256, 0, stream>>>(out_m, s_tmp, sm, (size_t)NN, CM);

        // ---- z = z + outer_product_mean(m) ----
        lnrun(stream, out_m, P.opm.ln, m_ln, NRES, CM);
        gemm(stream, m_ln, CM, 0, 0, P.opm.l1.w, 32, 0, 0, 0, P.opm.l1.b,
             a_op, 32, 0, 0, NRES, 32, CM, 1, 1);
        k_rowmask_mul<<<ewg(NRES * 32), 256, 0, stream>>>(a_op, sm, (size_t)NRES * 32, 32);
        gemm(stream, m_ln, CM, 0, 0, P.opm.l2.w, 32, 0, 0, 0, P.opm.l2.b,
             b_op, 32, 0, 0, NRES, 32, CM, 1, 1);
        k_rowmask_mul<<<ewg(NRES * 32), 256, 0, stream>>>(b_op, sm, (size_t)NRES * 32, 32);
        // T[c][j][z'] = b @ Wout[c*32:(c+1)*32, :]   (32 batches)
        gemm(stream, b_op, 32, 0, 0, P.opm.out.w, CZ, 32LL * CZ, 0, 0, nullptr,
             T_op, CZ, (long long)NRES * CZ, 0, NRES, CZ, 32, 32, 1);
        // outer[i][j][z'] = a @ T[:,j,:]   (256 batches over j), bias added once
        gemm(stream, a_op, 32, 0, 0, T_op, (long long)NRES * CZ, CZ, 0, 0, P.opm.out.b,
             bB, (long long)NRES * CZ, CZ, 0, NRES, CZ, 32, NRES, 1);
        k_opm_finalize<<<ewg(BIGF), 256, 0, stream>>>(out_z, bB, sm, BIGF);

        // ---- triangle multiplications ----
        trimul(stream, out_z, pm, P.tmo, true,  bA, bB, bC, bD, bE, bG);
        trimul(stream, out_z, pm, P.tmi, false, bA, bB, bC, bD, bE, bG);

        // ---- triangle attention (start / end) ----
        tri_attention(stream, out_z, out_z, pm, P.tas, bA, bB, bC, bD, bE, bF, bG, tb, false);
        k_transpose_pair<<<ewg(BIGF), 256, 0, stream>>>(bH, out_z, BIGF);
        k_transpose_mask<<<ewg(NN), 256, 0, stream>>>(pm_t, pm, (size_t)NN);
        tri_attention(stream, out_z, bH, pm_t, P.tae, bA, bB, bC, bD, bE, bF, bG, tb, true);

        // ---- z = z + transition(z)  (chunked over rows) ----
        lnrun(stream, out_z, P.pair_trans.ln, bA, NN, CZ);
        for (int r0 = 0; r0 < NN; r0 += 16384) {
            gemm(stream, bA + (long long)r0 * CZ, CZ, 0, 0, P.pair_trans.l1.w, 512, 0, 0, 0,
                 P.pair_trans.l1.b, bG, 512, 0, 0, 16384, 512, CZ, 1, 1, 1.f, 1, 0);
            gemm(stream, bG, 512, 0, 0, P.pair_trans.l2.w, CZ, 0, 0, 0,
                 P.pair_trans.l2.b, bB, CZ, 0, 0, 16384, CZ, 512, 1, 1);
            k_residual_rowmask<<<ewg((size_t)16384 * CZ), 256, 0, stream>>>(
                out_z + (long long)r0 * CZ, bB, pm + r0, (size_t)16384 * CZ, CZ);
        }
    }

    // ---- s = m @ Wf + bf ----
    gemm(stream, out_m, CM, 0, 0, fin_w, CS, 0, 0, 0, fin_b, out_s, CS, 0, 0,
         NRES, CS, CM, 1, 1);
}
